// TransformerBlock_128849018921
// MI455X (gfx1250) — matrix operand order repaired and checked by reading, not compiled, not run on hardware
//
#include <hip/hip_runtime.h>
#include <stdint.h>

typedef __attribute__((ext_vector_type(16))) __bf16 v16bf;
typedef __attribute__((ext_vector_type(8)))  float  v8f;
typedef __attribute__((ext_vector_type(4)))  int    v4i;

#define GLOBAL_AS __attribute__((address_space(1)))
#define LDS_AS    __attribute__((address_space(3)))

#ifndef __has_builtin
#define __has_builtin(x) 0
#endif

#if __has_builtin(__builtin_amdgcn_global_load_async_to_lds_b128)
#define HAVE_ASYNC_LDS 1
#else
#define HAVE_ASYNC_LDS 0
#endif

#if HAVE_ASYNC_LDS
__device__ __forceinline__ void async_copy16(const void* g, void* lds) {
  __builtin_amdgcn_global_load_async_to_lds_b128(
      (GLOBAL_AS v4i*)g, (LDS_AS v4i*)lds, 0, 0);
}
__device__ __forceinline__ void wait_async0() {
#if __has_builtin(__builtin_amdgcn_s_wait_asynccnt)
  __builtin_amdgcn_s_wait_asynccnt(0);
#else
  asm volatile("s_wait_asynccnt 0" ::: "memory");
#endif
}
#endif

union Frag { v16bf v; uint32_t u[8]; };

__device__ __forceinline__ unsigned short f32_to_bf16(float f) {
  union { float f; uint32_t u; } c; c.f = f;
  uint32_t u = c.u;
  u += 0x7FFFu + ((u >> 16) & 1u);   // round-to-nearest-even
  return (unsigned short)(u >> 16);
}

__device__ __forceinline__ v8f vzero8() {
  v8f z;
  #pragma unroll
  for (int e = 0; e < 8; ++e) z[e] = 0.0f;
  return z;
}

// reductions across the 16 lanes of a half-wave (C/D layout groups)
__device__ __forceinline__ float half16_max(float v) {
  v = fmaxf(v, __shfl_xor(v, 1, 32));
  v = fmaxf(v, __shfl_xor(v, 2, 32));
  v = fmaxf(v, __shfl_xor(v, 4, 32));
  v = fmaxf(v, __shfl_xor(v, 8, 32));
  return v;
}
__device__ __forceinline__ float half16_sum(float v) {
  v += __shfl_xor(v, 1, 32);
  v += __shfl_xor(v, 2, 32);
  v += __shfl_xor(v, 4, 32);
  v += __shfl_xor(v, 8, 32);
  return v;
}

// ---------------------------------------------------------------------------
// Weight convert+transpose: src f32 [K][N] -> dst bf16 [N][K]
// ---------------------------------------------------------------------------
__global__ void transpose_to_bf16(const float* __restrict__ src,
                                  unsigned short* __restrict__ dst,
                                  int K, int N) {
  size_t i = (size_t)blockIdx.x * 256 + threadIdx.x;
  size_t total = (size_t)K * N;
  if (i >= total) return;
  int n = (int)(i / K);
  int k = (int)(i % K);
  dst[i] = f32_to_bf16(src[(size_t)k * N + n]);
}

__global__ void concat_bias3(const float* __restrict__ a,
                             const float* __restrict__ b,
                             const float* __restrict__ c,
                             float* __restrict__ out) {
  int i = blockIdx.x * 256 + threadIdx.x;
  if (i < 768)        out[i] = a[i];
  else if (i < 1536)  out[i] = b[i - 768];
  else if (i < 2304)  out[i] = c[i - 1536];
}

// ---------------------------------------------------------------------------
// LayerNorm over D=768, one row per workgroup, bf16 output
// ---------------------------------------------------------------------------
__global__ __launch_bounds__(256) void layernorm_bf16_kernel(
    const float* __restrict__ x, const float* __restrict__ g,
    const float* __restrict__ bta, unsigned short* __restrict__ out) {
  const int D = 768;
  size_t row = blockIdx.x;
  const float* xr = x + row * D;
  int tid = threadIdx.x, lane = tid & 31, wv = tid >> 5;
  __shared__ float red[8];
  __shared__ float stat[2];

  float v0 = xr[tid], v1 = xr[tid + 256], v2 = xr[tid + 512];
  float s = v0 + v1 + v2;
  for (int m = 16; m; m >>= 1) s += __shfl_xor(s, m, 32);
  if (!lane) red[wv] = s;
  __syncthreads();
  if (tid == 0) {
    float t = 0; for (int i = 0; i < 8; ++i) t += red[i];
    stat[0] = t * (1.0f / 768.0f);
  }
  __syncthreads();
  float mu = stat[0];
  float d0 = v0 - mu, d1 = v1 - mu, d2 = v2 - mu;
  s = d0 * d0 + d1 * d1 + d2 * d2;
  for (int m = 16; m; m >>= 1) s += __shfl_xor(s, m, 32);
  if (!lane) red[wv] = s;
  __syncthreads();
  if (tid == 0) {
    float t = 0; for (int i = 0; i < 8; ++i) t += red[i];
    stat[1] = rsqrtf(t * (1.0f / 768.0f) + 1e-5f);
  }
  __syncthreads();
  float rs = stat[1];
  out[row * D + tid]       = f32_to_bf16(d0 * rs * g[tid]       + bta[tid]);
  out[row * D + tid + 256] = f32_to_bf16(d1 * rs * g[tid + 256] + bta[tid + 256]);
  out[row * D + tid + 512] = f32_to_bf16(d2 * rs * g[tid + 512] + bta[tid + 512]);
}

// ---------------------------------------------------------------------------
// Tiled bf16 WMMA GEMM: C[M][N] = A[M][K] * Bt[N][K]^T + bias (+ epilogue)
// OP 0: bf16 out, and pre-scales columns n<768 (Q block) by 1/sqrt(dk)=0.125
// OP 1: exact GELU -> bf16 out; OP 2: residual add -> f32 out
// Block 256 thr (8 waves), tile 128x128, BK=64; wave = 32x64 (2x4 WMMA tiles)
// LDS rows padded to 72 halfs: 144B = 16B-aligned rows (B128 transfers) and
// 36-dword stride -> 16 distinct banks across the 16 lanes of a half-wave.
// ---------------------------------------------------------------------------
template <int OP>
__global__ __launch_bounds__(256) void gemm_bf16_wmma(
    const unsigned short* __restrict__ A, const unsigned short* __restrict__ Bt,
    const float* __restrict__ bias, const float* __restrict__ res,
    void* __restrict__ outp, int M, int N, int K) {
  __shared__ __attribute__((aligned(16))) unsigned short As[128][72];
  __shared__ __attribute__((aligned(16))) unsigned short Bs[128][72];

  const int tid  = threadIdx.x;
  const int lane = tid & 31;
  const int wave = tid >> 5;
  const int hh   = lane >> 4;   // half-wave
  const int l16  = lane & 15;
  const int wRow = wave >> 1;   // 0..3 -> 32 rows each
  const int wCol = wave & 1;    // 0..1 -> 64 cols each

  const int m0 = blockIdx.y * 128;
  const int n0 = blockIdx.x * 128;

  const uint4* A128 = reinterpret_cast<const uint4*>(A);
  const uint4* B128 = reinterpret_cast<const uint4*>(Bt);
  const int Ku4 = K >> 3;  // uint4 (8 halfs) per row

  v8f acc[2][4];
  #pragma unroll
  for (int i = 0; i < 2; ++i)
    #pragma unroll
    for (int j = 0; j < 4; ++j) acc[i][j] = vzero8();

  for (int k0 = 0; k0 < K; k0 += 64) {
    #pragma unroll
    for (int i = 0; i < 2; ++i) {        // 128x32 halfs = 512 uint4 per operand
      int idx = tid + i * 256;           // 0..511
      int row = idx >> 2;                // 4 uint4 per row slab
      int c4  = idx & 3;
      size_t ga = (size_t)(m0 + row) * Ku4 + (k0 >> 3) + c4;
      size_t gb = (size_t)(n0 + row) * Ku4 + (k0 >> 3) + c4;
#if HAVE_ASYNC_LDS
      async_copy16(&A128[ga], &As[row][c4 * 8]);
      async_copy16(&B128[gb], &Bs[row][c4 * 8]);
#else
      uint4 a = A128[ga];
      *reinterpret_cast<uint4*>(&As[row][c4 * 8]) = a;
      uint4 b = B128[gb];
      *reinterpret_cast<uint4*>(&Bs[row][c4 * 8]) = b;
#endif
      if (k0 + 64 < K) {                 // prefetch next k-tile (global_prefetch_b8)
        __builtin_prefetch(&A128[ga + 8], 0, 1);
        __builtin_prefetch(&B128[gb + 8], 0, 1);
      }
    }
#if HAVE_ASYNC_LDS
    wait_async0();
#endif
    __syncthreads();

    #pragma unroll
    for (int ks = 0; ks < 2; ++ks) {
      Frag af[2], bfg[4];
      #pragma unroll
      for (int mt = 0; mt < 2; ++mt) {
        int m = wRow * 32 + mt * 16 + l16;
        #pragma unroll
        for (int v = 0; v < 8; ++v) {
          int k = ks * 32 + ((v >= 4) ? 16 : 0) + hh * 8 + (v & 3) * 2;
          af[mt].u[v] = *reinterpret_cast<const uint32_t*>(&As[m][k]);
        }
      }
      #pragma unroll
      for (int nt = 0; nt < 4; ++nt) {
        int n = wCol * 64 + nt * 16 + l16;
        #pragma unroll
        for (int v = 0; v < 8; ++v) {
          int k = ks * 32 + ((v >= 4) ? 16 : 0) + hh * 8 + (v & 3) * 2;
          bfg[nt].u[v] = *reinterpret_cast<const uint32_t*>(&Bs[n][k]);
        }
      }
      #pragma unroll
      for (int mt = 0; mt < 2; ++mt)
        #pragma unroll
        for (int nt = 0; nt < 4; ++nt)
          acc[mt][nt] = __builtin_amdgcn_wmma_f32_16x16x32_bf16(
              false, af[mt].v, false, bfg[nt].v, (short)0, acc[mt][nt],
              false, false);
    }
    __syncthreads();
  }

  #pragma unroll
  for (int mt = 0; mt < 2; ++mt) {
    #pragma unroll
    for (int nt = 0; nt < 4; ++nt) {
      int n = n0 + wCol * 64 + nt * 16 + l16;
      float bn = bias[n];
      #pragma unroll
      for (int r = 0; r < 8; ++r) {
        int m = m0 + wRow * 32 + mt * 16 + hh * 8 + r;
        float vv = acc[mt][nt][r] + bn;
        size_t o = (size_t)m * N + n;
        if (OP == 0) {
          if (n < 768) vv *= 0.125f;  // fold 1/sqrt(dk) into Q
          ((unsigned short*)outp)[o] = f32_to_bf16(vv);
        } else if (OP == 1) {
          float gl = 0.5f * vv * (1.0f + erff(vv * 0.70710678118654752f));
          ((unsigned short*)outp)[o] = f32_to_bf16(gl);
        } else {
          ((float*)outp)[o] = res[o] + vv;
        }
      }
    }
  }
}

// ---------------------------------------------------------------------------
// Flash attention: qkv bf16 [S][2304] (Q pre-scaled) -> ctx bf16 [S][768]
// Block: 128 queries, 8 waves (16 q-rows each); loop over 64-key blocks.
// ---------------------------------------------------------------------------
__global__ __launch_bounds__(256) void attention_wmma(
    const unsigned short* __restrict__ qkv, unsigned short* __restrict__ ctx,
    int S) {
  __shared__ __attribute__((aligned(16))) unsigned short Qs[128][72]; // [q][dk]
  __shared__ __attribute__((aligned(16))) unsigned short Ks[64][72];  // [key][dk]
  __shared__ __attribute__((aligned(16))) unsigned short Vs[64][72];  // [dk][key]
  __shared__ __attribute__((aligned(16))) unsigned short Ps[8][16][72]; // [q][key]

  const int tid  = threadIdx.x;
  const int lane = tid & 31;
  const int wv   = tid >> 5;
  const int hh   = lane >> 4;
  const int l16  = lane & 15;

  const int head = blockIdx.y;
  const int q0   = blockIdx.x * 128;

  const uint4* G4 = reinterpret_cast<const uint4*>(qkv);
  const int rowU4 = 288;  // 2304/8 uint4 per qkv row

  #pragma unroll
  for (int i = 0; i < 2; ++i) {  // Q tile 128x32 halfs = 512 uint4
    int idx = tid + i * 256;
    int r = idx >> 2;
    int c4 = idx & 3;
    const uint4* gq = &G4[(size_t)(q0 + r) * rowU4 + head * 8 + c4];
#if HAVE_ASYNC_LDS
    async_copy16(gq, &Qs[r][c4 * 8]);
#else
    uint4 d = *gq;
    *reinterpret_cast<uint4*>(&Qs[r][c4 * 8]) = d;
#endif
  }
#if HAVE_ASYNC_LDS
  wait_async0();
#endif
  __syncthreads();

  Frag qa[2];
  #pragma unroll
  for (int ks = 0; ks < 2; ++ks) {
    int m = wv * 16 + l16;
    #pragma unroll
    for (int v = 0; v < 8; ++v) {
      int k = ks * 32 + ((v >= 4) ? 16 : 0) + hh * 8 + (v & 3) * 2;
      qa[ks].u[v] = *reinterpret_cast<const uint32_t*>(&Qs[m][k]);
    }
  }

  float mrow[8], lrow[8];
  v8f O[4];
  #pragma unroll
  for (int r = 0; r < 8; ++r) { mrow[r] = -1e30f; lrow[r] = 0.0f; }
  #pragma unroll
  for (int t = 0; t < 4; ++t) O[t] = vzero8();

  const int nKB = S / 64;
  for (int kb = 0; kb < nKB; ++kb) {
    __syncthreads();  // previous block fully consumed
    {
      // K and V blocks: 64x32 halfs = 256 uint4 each; 1 per thread
      int r  = tid >> 2;
      int c4 = tid & 3;
      size_t base = (size_t)(kb * 64 + r) * rowU4 + head * 8 + c4;
#if HAVE_ASYNC_LDS
      async_copy16(&G4[base + 96], &Ks[r][c4 * 8]);   // K: +768 halfs
#else
      uint4 kd = G4[base + 96];
      *reinterpret_cast<uint4*>(&Ks[r][c4 * 8]) = kd;
#endif
      uint4 vd = G4[base + 192];  // V: +1536 halfs (manual: transposed store)
      int d0 = c4 * 8;
      Vs[d0 + 0][r] = (unsigned short)(vd.x & 0xFFFFu);
      Vs[d0 + 1][r] = (unsigned short)(vd.x >> 16);
      Vs[d0 + 2][r] = (unsigned short)(vd.y & 0xFFFFu);
      Vs[d0 + 3][r] = (unsigned short)(vd.y >> 16);
      Vs[d0 + 4][r] = (unsigned short)(vd.z & 0xFFFFu);
      Vs[d0 + 5][r] = (unsigned short)(vd.z >> 16);
      Vs[d0 + 6][r] = (unsigned short)(vd.w & 0xFFFFu);
      Vs[d0 + 7][r] = (unsigned short)(vd.w >> 16);
      if (kb + 1 < nKB) {  // prefetch next key block
        __builtin_prefetch(&G4[base + (size_t)64 * rowU4 + 96], 0, 1);
        __builtin_prefetch(&G4[base + (size_t)64 * rowU4 + 192], 0, 1);
      }
    }
#if HAVE_ASYNC_LDS
    wait_async0();
#endif
    __syncthreads();

    // S = Q K^T   (already includes 1/sqrt(dk) via pre-scaled Q)
    v8f sv[4];
    #pragma unroll
    for (int nt = 0; nt < 4; ++nt) {
      sv[nt] = vzero8();
      #pragma unroll
      for (int ks = 0; ks < 2; ++ks) {
        Frag kf;
        int n = nt * 16 + l16;
        #pragma unroll
        for (int v = 0; v < 8; ++v) {
          int k = ks * 32 + ((v >= 4) ? 16 : 0) + hh * 8 + (v & 3) * 2;
          kf.u[v] = *reinterpret_cast<const uint32_t*>(&Ks[n][k]);
        }
        sv[nt] = __builtin_amdgcn_wmma_f32_16x16x32_bf16(
            false, qa[ks].v, false, kf.v, (short)0, sv[nt], false, false);
      }
    }

    // online softmax (rows live per half-wave: m = hh*8 + r)
    #pragma unroll
    for (int r = 0; r < 8; ++r) {
      float mx = fmaxf(fmaxf(sv[0][r], sv[1][r]), fmaxf(sv[2][r], sv[3][r]));
      mx = half16_max(mx);
      float mnew  = fmaxf(mrow[r], mx);
      float alpha = __expf(mrow[r] - mnew);
      mrow[r] = mnew;
      float rs = 0.0f;
      #pragma unroll
      for (int nt = 0; nt < 4; ++nt) {
        float p = __expf(sv[nt][r] - mnew);
        sv[nt][r] = p;
        rs += p;
        Ps[wv][hh * 8 + r][nt * 16 + l16] = f32_to_bf16(p);
      }
      rs = half16_sum(rs);
      lrow[r] = lrow[r] * alpha + rs;
      #pragma unroll
      for (int t = 0; t < 4; ++t) O[t][r] *= alpha;
    }

    // O += P V   (DS ops are in-order within a wave; Ps is per-wave)
    #pragma unroll
    for (int ks = 0; ks < 2; ++ks) {
      Frag pa;
      #pragma unroll
      for (int v = 0; v < 8; ++v) {
        int k = ks * 32 + ((v >= 4) ? 16 : 0) + hh * 8 + (v & 3) * 2;
        pa.u[v] = *reinterpret_cast<const uint32_t*>(&Ps[wv][l16][k]);
      }
      #pragma unroll
      for (int nt = 0; nt < 4; ++nt) {
        Frag vf;
        int n = nt * 16 + l16;
        #pragma unroll
        for (int v = 0; v < 8; ++v) {
          int k = ks * 32 + ((v >= 4) ? 16 : 0) + hh * 8 + (v & 3) * 2;
          vf.u[v] = *reinterpret_cast<const uint32_t*>(&Vs[n][k]);
        }
        O[nt] = __builtin_amdgcn_wmma_f32_16x16x32_bf16(
            false, pa.v, false, vf.v, (short)0, O[nt], false, false);
      }
    }
  }

  #pragma unroll
  for (int nt = 0; nt < 4; ++nt) {
    int col = head * 64 + nt * 16 + l16;
    #pragma unroll
    for (int r = 0; r < 8; ++r) {
      int row = q0 + wv * 16 + hh * 8 + r;
      float vvv = O[nt][r] / lrow[r];
      ctx[(size_t)row * 768 + col] = f32_to_bf16(vvv);
    }
  }
}

// ---------------------------------------------------------------------------
extern "C" void kernel_launch(void* const* d_in, const int* in_sizes, int n_in,
                              void* d_out, int out_size, void* d_ws,
                              size_t ws_size, hipStream_t stream) {
  (void)in_sizes; (void)n_in; (void)out_size; (void)ws_size;
  const float* x    = (const float*)d_in[0];
  const float* Wq   = (const float*)d_in[1];
  const float* bq   = (const float*)d_in[2];
  const float* Wk   = (const float*)d_in[3];
  const float* bk   = (const float*)d_in[4];
  const float* Wv   = (const float*)d_in[5];
  const float* bv   = (const float*)d_in[6];
  const float* Wo   = (const float*)d_in[7];
  const float* bo   = (const float*)d_in[8];
  const float* ln1g = (const float*)d_in[9];
  const float* ln1b = (const float*)d_in[10];
  const float* ln2g = (const float*)d_in[11];
  const float* ln2b = (const float*)d_in[12];
  const float* W1   = (const float*)d_in[13];
  const float* b1   = (const float*)d_in[14];
  const float* W2   = (const float*)d_in[15];
  const float* b2   = (const float*)d_in[16];

  const int S = 4096, D = 768, F = 3072;

  char* ws = (char*)d_ws;
  size_t off = 0;
  auto alloc = [&](size_t bytes) -> char* {
    char* p = ws + off;
    off += (bytes + 255) & ~(size_t)255;
    return p;
  };

  unsigned short* WqkvT = (unsigned short*)alloc((size_t)3 * D * D * 2);
  unsigned short* WoT   = (unsigned short*)alloc((size_t)D * D * 2);
  unsigned short* W1T   = (unsigned short*)alloc((size_t)F * D * 2);
  unsigned short* W2T   = (unsigned short*)alloc((size_t)D * F * 2);
  float*          bqkv  = (float*)alloc((size_t)3 * D * 4);
  unsigned short* xn1   = (unsigned short*)alloc((size_t)S * D * 2);
  unsigned short* qkv   = (unsigned short*)alloc((size_t)S * 3 * D * 2);
  unsigned short* ctx   = (unsigned short*)alloc((size_t)S * D * 2);
  float*          x2    = (float*)alloc((size_t)S * D * 4);
  unsigned short* xn2   = (unsigned short*)alloc((size_t)S * D * 2);
  unsigned short* hbuf  = (unsigned short*)alloc((size_t)S * F * 2);

  dim3 blk(256);
  int gDD = (D * D + 255) / 256;
  transpose_to_bf16<<<gDD, blk, 0, stream>>>(Wq, WqkvT, D, D);
  transpose_to_bf16<<<gDD, blk, 0, stream>>>(Wk, WqkvT + (size_t)D * D, D, D);
  transpose_to_bf16<<<gDD, blk, 0, stream>>>(Wv, WqkvT + (size_t)2 * D * D, D, D);
  transpose_to_bf16<<<gDD, blk, 0, stream>>>(Wo, WoT, D, D);
  int gDF = (D * F + 255) / 256;
  transpose_to_bf16<<<gDF, blk, 0, stream>>>(W1, W1T, D, F);  // [768][3072] -> [3072][768]
  transpose_to_bf16<<<gDF, blk, 0, stream>>>(W2, W2T, F, D);  // [3072][768] -> [768][3072]
  concat_bias3<<<9, blk, 0, stream>>>(bq, bk, bv, bqkv);

  // xn1 = LN1(x)  (bf16)
  layernorm_bf16_kernel<<<S, blk, 0, stream>>>(x, ln1g, ln1b, xn1);

  // qkv = xn1 @ [Wq|Wk|Wv] + [bq|bk|bv]  (bf16; Q block pre-scaled by 0.125)
  gemm_bf16_wmma<0><<<dim3((3 * D) / 128, S / 128), blk, 0, stream>>>(
      xn1, WqkvT, bqkv, nullptr, qkv, S, 3 * D, D);

  // ctx = softmax(Q K^T) V  (bf16)
  attention_wmma<<<dim3(S / 128, 12), blk, 0, stream>>>(qkv, ctx, S);

  // x2 = x + ctx @ Wo + bo  (f32)
  gemm_bf16_wmma<2><<<dim3(D / 128, S / 128), blk, 0, stream>>>(
      ctx, WoT, bo, x, x2, S, D, D);

  // xn2 = LN2(x2)  (bf16)
  layernorm_bf16_kernel<<<S, blk, 0, stream>>>(x2, ln2g, ln2b, xn2);

  // hbuf = gelu(xn2 @ W1 + b1)  (bf16)
  gemm_bf16_wmma<1><<<dim3(F / 128, S / 128), blk, 0, stream>>>(
      xn2, W1T, b1, nullptr, hbuf, S, F, D);

  // out = x2 + hbuf @ W2 + b2  (f32)
  gemm_bf16_wmma<2><<<dim3(D / 128, S / 128), blk, 0, stream>>>(
      hbuf, W2T, b2, x2, (float*)d_out, S, D, F);
}
